// QuoraModel_74217034875485
// MI455X (gfx1250) — matrix operand order repaired and checked by reading, not compiled, not run on hardware
//
#include <hip/hip_runtime.h>
#include <hip/hip_bf16.h>
#include <cstdint>

// ---------------------------------------------------------------------------
// Model dims
// ---------------------------------------------------------------------------
constexpr int kB  = 256;          // batch
constexpr int kT  = 50;           // seq len
constexpr int kE  = 300;          // embed dim
constexpr int kEp = 320;          // embed dim padded to K%32==0
constexpr int kH  = 256;          // hidden
constexpr int kG  = 1024;         // 4*H (LSTM gates)
constexpr int kBT = kB * kT;      // 12800 flattened rows

typedef unsigned short u16;
typedef __attribute__((ext_vector_type(16))) __bf16 v16bf;
typedef __attribute__((ext_vector_type(8)))  __bf16 v8bf;
typedef __attribute__((ext_vector_type(8)))  float  v8f;

// ---------------------------------------------------------------------------
// bf16 helpers (RNE float->bf16, bit reinterpret)
// ---------------------------------------------------------------------------
__device__ __forceinline__ u16 f2bf(float f) {
  union { float f; unsigned u; } v; v.f = f;
  unsigned r = v.u + 0x7FFFu + ((v.u >> 16) & 1u);
  return (u16)(r >> 16);
}
__device__ __forceinline__ __bf16 bfbits(u16 u) {
  union { u16 u; __bf16 b; } v; v.u = u; return v.b;
}
__device__ __forceinline__ v16bf mk16(v8bf lo, v8bf hi) {
  v16bf r;
#pragma unroll
  for (int i = 0; i < 8; ++i) { r[i] = lo[i]; r[i + 8] = hi[i]; }
  return r;
}
__device__ __forceinline__ v16bf zero16() {
  v16bf r;
#pragma unroll
  for (int i = 0; i < 16; ++i) r[i] = bfbits(0);
  return r;
}

// A-matrix fragment (16x32 bf16). lane row m = lane&15,
// elem i -> K = kc*32 + half*16 + i (fully contiguous 16).
// rowbase points at A row m (caller supplies per-lane row base).
__device__ __forceinline__ v16bf load_arow(const u16* rowbase, int kc) {
  int half = (threadIdx.x >> 4) & 1;
  const u16* p = rowbase + kc * 32 + half * 16;
  v8bf lo = *(const v8bf*)p;
  v8bf hi = *(const v8bf*)(p + 8);
  return mk16(lo, hi);
}
// B-matrix fragment (32x16 bf16) from a contiguous-K source (a row of B^T):
// lane col n = lane&15 ; elem i -> K = kc*32 + half*16 + i (fully contiguous 16).
__device__ __forceinline__ v16bf load_brow(const u16* rowbase, int kc) {
  int half = (threadIdx.x >> 4) & 1;
  const u16* p = rowbase + kc * 32 + half * 16;
  v8bf lo = *(const v8bf*)p;
  v8bf hi = *(const v8bf*)(p + 8);
  return mk16(lo, hi);
}
__device__ __forceinline__ v8f wmma_bf16(v16bf a, v16bf b, v8f c) {
  return __builtin_amdgcn_wmma_f32_16x16x32_bf16(false, a, false, b, (short)0, c,
                                                 false, false);
}
__device__ __forceinline__ float sigm(float x) { return 1.0f / (1.0f + __expf(-x)); }

// ---------------------------------------------------------------------------
// Kernel: pack fp32 weights -> bf16 row-major with zero row-padding (K pad)
// ---------------------------------------------------------------------------
__global__ void pack_bf16(const float* __restrict__ src, u16* __restrict__ dst,
                          int rows, int cols, int rowsPad) {
  int total = rowsPad * cols;
  for (int idx = blockIdx.x * blockDim.x + threadIdx.x; idx < total;
       idx += gridDim.x * blockDim.x) {
    int r = idx / cols;
    dst[idx] = (r < rows) ? f2bf(src[idx]) : (u16)0;
  }
}

// ---------------------------------------------------------------------------
// Kernel: pack fp32 weights (KxN row-major) -> bf16 TRANSPOSED (NxK).
// Makes recurrence B-fragments two contiguous b128 loads.
// ---------------------------------------------------------------------------
__global__ void pack_bf16_T(const float* __restrict__ src, u16* __restrict__ dst,
                            int K, int N) {
  int total = K * N;
  for (int idx = blockIdx.x * blockDim.x + threadIdx.x; idx < total;
       idx += gridDim.x * blockDim.x) {
    int k = idx / N, n = idx - k * N;
    dst[(size_t)n * K + k] = f2bf(src[idx]);
  }
}

// ---------------------------------------------------------------------------
// Kernel: embedding gather + bf16 convert + zero pad cols 300..319
// ---------------------------------------------------------------------------
__global__ void gather_embed(const int* __restrict__ tokens,
                             const float* __restrict__ emb, u16* __restrict__ dst) {
  const int total = kBT * kEp;
  for (int idx = blockIdx.x * blockDim.x + threadIdx.x; idx < total;
       idx += gridDim.x * blockDim.x) {
    int row = idx / kEp, col = idx - row * kEp;
    int tok = tokens[row];
    dst[idx] = (col < kE) ? f2bf(emb[(size_t)tok * kE + col]) : (u16)0;
  }
}

// ---------------------------------------------------------------------------
// Kernel: generic bf16 WMMA GEMM   C(MxN) = A(MxK) * B(KxN) + bias
// 256 threads = 8 waves; wave w -> M-tile blockIdx.y*8+w ; N-tile = blockIdx.x.
// Double-buffered LDS staging of the 32x16 B tile: the next tile's global
// load is issued before the current WMMA, hiding the load behind compute;
// one barrier per K-chunk.
// ---------------------------------------------------------------------------
__global__ __launch_bounds__(256) void gemm_bf16(
    const u16* __restrict__ A, const u16* __restrict__ Bw,
    const float* __restrict__ bias, float* __restrict__ Cf, u16* __restrict__ Cbf,
    int M, int N, int K, int lda, int ldb) {
  __shared__ u16 bsT[2][16 * 32];  // [buf][n][k]
  int tid = threadIdx.x, wave = tid >> 5, lane = tid & 31;
  int nl = lane & 15, half = (lane >> 4) & 1;
  int mt = blockIdx.y * 8 + wave;
  int m0 = mt * 16, n0 = blockIdx.x * 16;
  // static staging map: row k = tid>>3 (0..31), col pair n2 = (tid&7)*2
  int sk = tid >> 3, sn = (tid & 7) * 2;
  const u16* srow = Bw + (size_t)sk * ldb + n0 + sn;
  const u16* arow = A + (size_t)(m0 + nl) * lda;
  int nkc = K >> 5;
  // prologue: stage tile 0 into buffer 0
  {
    unsigned pv = *(const unsigned*)srow;
    bsT[0][sn * 32 + sk] = (u16)(pv & 0xFFFFu);
    bsT[0][(sn + 1) * 32 + sk] = (u16)(pv >> 16);
  }
  __syncthreads();
  v8f acc = {};
  for (int kc = 0; kc < nkc; ++kc) {
    int cur = kc & 1;
    bool more = (kc + 1 < nkc);
    unsigned pv = 0;
    if (more) pv = *(const unsigned*)(srow + (size_t)(kc + 1) * 32 * ldb);
    if (kc + 2 < nkc) __builtin_prefetch(arow + (kc + 2) * 32, 0, 1);
    v16bf a = load_arow(arow, kc);
    v16bf b = load_brow(&bsT[cur][nl * 32], 0);  // tile-relative K offset
    acc = wmma_bf16(a, b, acc);
    if (more) {
      int nxt = cur ^ 1;
      bsT[nxt][sn * 32 + sk] = (u16)(pv & 0xFFFFu);
      bsT[nxt][(sn + 1) * 32 + sk] = (u16)(pv >> 16);
    }
    __syncthreads();
  }
  int n = n0 + nl;
  float bv = bias ? bias[n] : 0.0f;
#pragma unroll
  for (int r = 0; r < 8; ++r) {
    int m = m0 + r + 8 * half;
    float v = acc[r] + bv;
    if (Cf)  Cf[(size_t)m * N + n] = v;
    if (Cbf) Cbf[(size_t)m * N + n] = f2bf(v);
  }
}

// ---------------------------------------------------------------------------
// Kernel: tanh-RNN recurrence.  h_t = tanh(xw_t + h_{t-1} @ Wh)
// Block owns 16 batch rows; h double-buffered in LDS (bf16) -> one barrier
// per step; WhT (transposed bf16) gives two-b128 B fragments, L2-resident.
// blockIdx.y selects sequence {A,B}.  Output masked; state stays unmasked.
// ---------------------------------------------------------------------------
__global__ __launch_bounds__(256) void rnn_rec(
    const float* __restrict__ xwA, const float* __restrict__ xwB,
    const u16* __restrict__ WhT, const int* __restrict__ maskA,
    const int* __restrict__ maskB, u16* __restrict__ outA, u16* __restrict__ outB) {
  const float* xw  = blockIdx.y ? xwB : xwA;
  const int*  mask = blockIdx.y ? maskB : maskA;
  u16*       outbf = blockIdx.y ? outB : outA;

  __shared__ u16 hbuf[2][16 * kH];
  int tid = threadIdx.x, wave = tid >> 5, lane = tid & 31;
  int nl = lane & 15, half = (lane >> 4) & 1;
  int b0 = blockIdx.x * 16;
  for (int i = tid; i < 16 * kH; i += 256) hbuf[0][i] = 0;
  __syncthreads();

  // per-lane B^T row bases for this wave's two N-tiles
  const u16* bt0 = WhT + (size_t)((wave * 2 + 0) * 16 + nl) * kH;
  const u16* bt1 = WhT + (size_t)((wave * 2 + 1) * 16 + nl) * kH;

  for (int t = 0; t < kT; ++t) {
    const u16* hr = hbuf[t & 1];        // h_{t-1}
    u16* hw = hbuf[(t & 1) ^ 1];        // h_t
    v8f acc0 = {}, acc1 = {};
    for (int kc = 0; kc < 8; ++kc) {
      v16bf a = load_arow(hr + nl * kH, kc);
      acc0 = wmma_bf16(a, load_brow(bt0, kc), acc0);
      acc1 = wmma_bf16(a, load_brow(bt1, kc), acc1);
    }
#pragma unroll
    for (int j = 0; j < 2; ++j) {
      v8f& acc = j ? acc1 : acc0;
      int n = (wave * 2 + j) * 16 + nl;
#pragma unroll
      for (int r = 0; r < 8; ++r) {
        int m = r + 8 * half;
        int row = b0 + m;
        float pre = acc[r] + xw[((size_t)row * kT + t) * kH + n];
        if (t + 1 < kT)
          __builtin_prefetch(&xw[((size_t)row * kT + t + 1) * kH + n], 0, 1);
        float h = tanhf(pre);
        hw[m * kH + n] = f2bf(h);
        float hm = (mask == nullptr || mask[row * kT + t]) ? h : 0.0f;
        outbf[((size_t)row * kT + t) * kH + n] = f2bf(hm);
      }
    }
    __syncthreads();  // h_t visible for next step
  }
}

// ---------------------------------------------------------------------------
// Kernel: fused coattention per batch item.
//   L = e1q @ e2^T (50x50 padded to 64x64), softmax rows, ctx = att @ e2,
//   enc = [e1q, ctx] written in bf16.  e2^T staged once in LDS so phase-3
//   B-fragments are two ds_load_b128.
// ---------------------------------------------------------------------------
__global__ __launch_bounds__(256) void coattn(const u16* __restrict__ e1q,
                                              const u16* __restrict__ e2,
                                              u16* __restrict__ encbf) {
  __shared__ float Lbuf[64 * 64];   // 16 KB
  __shared__ u16 attb[64 * 64];     // 8 KB
  __shared__ u16 e2T[kH * 64];      // 32 KB  [h][s]
  int b = blockIdx.x;
  int tid = threadIdx.x, wave = tid >> 5, lane = tid & 31;
  int nl = lane & 15, half = (lane >> 4) & 1;
  const u16* e1b = e1q + (size_t)b * kT * kH;
  const u16* e2b = e2 + (size_t)b * kT * kH;

  // zero-fill e2T (covers s>=50 pad), then transpose-copy e2 into it
  for (int idx = tid; idx < kH * 64; idx += 256) e2T[idx] = 0;
  __syncthreads();
  for (int idx = tid; idx < kT * kH; idx += 256) {
    int s = idx >> 8, h = idx & 255;            // contiguous global reads
    e2T[h * 64 + s] = e2b[idx];
  }
  // enc[:, 0:256] = e1q  (independent of LDS)
  for (int idx = tid; idx < kT * kH; idx += 256) {
    int trow = idx >> 8, col = idx & 255;
    encbf[((size_t)b * kT + trow) * 512 + col] = e1b[trow * kH + col];
  }

  // Phase 1: affinity L (4x4 tiles of 16x16, K = 256). e2^T columns are
  // contiguous-K rows of e2 -> free transpose via load_brow on e2 itself.
#pragma unroll
  for (int jt = 0; jt < 2; ++jt) {
    int tile = wave * 2 + jt;
    int mt = tile >> 2, nt = tile & 3;
    int tm = mt * 16 + nl;  // t index (A row)
    int sn = nt * 16 + nl;  // s index (B col)
    v8f acc = {};
    for (int kc = 0; kc < 8; ++kc) {
      v16bf a = (tm < kT) ? load_arow(e1b + tm * kH, kc) : zero16();
      v16bf bf = (sn < kT) ? load_brow(e2b + sn * kH, kc) : zero16();
      acc = wmma_bf16(a, bf, acc);
    }
#pragma unroll
    for (int r = 0; r < 8; ++r)
      Lbuf[(mt * 16 + r + 8 * half) * 64 + nt * 16 + nl] = acc[r];
  }
  __syncthreads();   // Lbuf ready; e2T copy also complete

  // Phase 2: row softmax over valid s<50; pad rows/cols -> 0 in att.
  if (tid < 64) {
    int row = tid;
    if (row < kT) {
      float mx = -1e30f;
      for (int s = 0; s < kT; ++s) mx = fmaxf(mx, Lbuf[row * 64 + s]);
      float sum = 0.0f;
      for (int s = 0; s < kT; ++s) sum += __expf(Lbuf[row * 64 + s] - mx);
      float inv = 1.0f / sum;
      for (int s = 0; s < 64; ++s) {
        float v = (s < kT) ? __expf(Lbuf[row * 64 + s] - mx) * inv : 0.0f;
        attb[row * 64 + s] = f2bf(v);
      }
    } else {
      for (int s = 0; s < 64; ++s) attb[row * 64 + s] = 0;
    }
  }
  __syncthreads();

  // Phase 3: ctx = att(64x64) @ e2(64x256); B from LDS e2T, two b128 each.
#pragma unroll
  for (int jj = 0; jj < 8; ++jj) {
    int tile = wave * 8 + jj;
    int mt = tile >> 4, nt = tile & 15;
    int trow_a = mt * 16 + nl;
    v8f acc = {};
    for (int kc = 0; kc < 2; ++kc) {
      v16bf a = load_arow(attb + trow_a * 64, kc);
      v16bf bf = load_brow(e2T + (nt * 16 + nl) * 64, kc);
      acc = wmma_bf16(a, bf, acc);
    }
#pragma unroll
    for (int r = 0; r < 8; ++r) {
      int trow = mt * 16 + r + 8 * half;
      if (trow < kT)
        encbf[((size_t)b * kT + trow) * 512 + 256 + nt * 16 + nl] = f2bf(acc[r]);
    }
  }
}

// ---------------------------------------------------------------------------
// Kernel: LSTM recurrence (one direction per blockIdx.y).
// Each wave owns hidden slice [wave*32, wave*32+32): 8 gate accumulators and
// the cell state c live in registers; h double-buffered in LDS (one barrier
// per step); U transposed -> two-b128 B fragments.
// ---------------------------------------------------------------------------
__global__ __launch_bounds__(256) void lstm_rec(
    const float* __restrict__ xwF, const u16* __restrict__ UfT,
    float* __restrict__ hF, const float* __restrict__ xwB,
    const u16* __restrict__ UbT, float* __restrict__ hB) {
  const float* xw = blockIdx.y ? xwB : xwF;
  const u16* UT = blockIdx.y ? UbT : UfT;
  float* hout = blockIdx.y ? hB : hF;
  const int reverse = blockIdx.y;

  __shared__ u16 hbuf[2][16 * kH];
  int tid = threadIdx.x, wave = tid >> 5, lane = tid & 31;
  int nl = lane & 15, half = (lane >> 4) & 1;
  int b0 = blockIdx.x * 16;
  int hc = wave * 32;
  for (int i = tid; i < 16 * kH; i += 256) hbuf[0][i] = 0;
  v8f c0 = {}, c1 = {};
  __syncthreads();

  // per-lane B^T row bases: gate g, sub-tile j -> column g*kH + hc + j*16 + nl
  const u16* btab[4][2];
#pragma unroll
  for (int g = 0; g < 4; ++g)
#pragma unroll
    for (int j = 0; j < 2; ++j)
      btab[g][j] = UT + (size_t)(g * kH + hc + j * 16 + nl) * kH;

  for (int s = 0; s < kT; ++s) {
    int t = reverse ? (kT - 1 - s) : s;
    const u16* hr = hbuf[s & 1];
    u16* hw = hbuf[(s & 1) ^ 1];
    v8f zero = {};
    v8f acc[4][2];
#pragma unroll
    for (int g = 0; g < 4; ++g) { acc[g][0] = zero; acc[g][1] = zero; }

    for (int kc = 0; kc < 8; ++kc) {
      v16bf a = load_arow(hr + nl * kH, kc);
#pragma unroll
      for (int g = 0; g < 4; ++g) {
#pragma unroll
        for (int j = 0; j < 2; ++j)
          acc[g][j] = wmma_bf16(a, load_brow(btab[g][j], kc), acc[g][j]);
      }
    }
#pragma unroll
    for (int j = 0; j < 2; ++j) {
      int n = hc + j * 16 + nl;
      v8f& cc = j ? c1 : c0;
#pragma unroll
      for (int r = 0; r < 8; ++r) {
        int m = r + 8 * half, row = b0 + m;
        const float* xr = xw + ((size_t)row * kT + t) * kG;
        if (s + 1 < kT) {
          int tn = reverse ? (t - 1) : (t + 1);
          __builtin_prefetch(xw + ((size_t)row * kT + tn) * kG + n, 0, 1);
        }
        float iv = acc[0][j][r] + xr[n];
        float fv = acc[1][j][r] + xr[kH + n];
        float gv = acc[2][j][r] + xr[2 * kH + n];
        float ov = acc[3][j][r] + xr[3 * kH + n];
        float cn = sigm(fv) * cc[r] + sigm(iv) * tanhf(gv);
        cc[r] = cn;
        float h = sigm(ov) * tanhf(cn);
        hw[m * kH + n] = f2bf(h);
        hout[((size_t)row * kT + t) * kH + n] = h;
      }
    }
    __syncthreads();
  }
}

// ---------------------------------------------------------------------------
// Kernel: classifier  logits[b] = flatten([hf,hb])[b,:] @ Wc + bc  (K=25600,C=2)
// ---------------------------------------------------------------------------
__global__ __launch_bounds__(256) void classify(const float* __restrict__ hf,
                                                const float* __restrict__ hb,
                                                const float* __restrict__ Wc,
                                                const float* __restrict__ bc,
                                                float* __restrict__ out) {
  __shared__ float red[512];
  int b = blockIdx.x, tid = threadIdx.x;
  float p0 = 0.0f, p1 = 0.0f;
  for (int k = tid; k < kT * 512; k += 256) {
    int t = k >> 9, j = k & 511;
    float v = (j < 256) ? hf[((size_t)b * kT + t) * kH + j]
                        : hb[((size_t)b * kT + t) * kH + (j - 256)];
    p0 += v * Wc[(size_t)k * 2 + 0];
    p1 += v * Wc[(size_t)k * 2 + 1];
  }
  red[tid] = p0;
  red[256 + tid] = p1;
  __syncthreads();
  for (int off = 128; off > 0; off >>= 1) {
    if (tid < off) {
      red[tid] += red[tid + off];
      red[256 + tid] += red[256 + tid + off];
    }
    __syncthreads();
  }
  if (tid == 0) {
    out[b * 2 + 0] = red[0] + bc[0];
    out[b * 2 + 1] = red[256] + bc[1];
  }
}

// ---------------------------------------------------------------------------
// Host launcher
// ---------------------------------------------------------------------------
extern "C" void kernel_launch(void* const* d_in, const int* in_sizes, int n_in,
                              void* d_out, int out_size, void* d_ws, size_t ws_size,
                              hipStream_t stream) {
  (void)in_sizes; (void)n_in; (void)out_size; (void)ws_size;
  const int*   tokens1 = (const int*)d_in[0];
  const int*   tokens2 = (const int*)d_in[1];
  const int*   mask1   = (const int*)d_in[2];
  const int*   mask2   = (const int*)d_in[3];
  const float* emb     = (const float*)d_in[4];
  const float* Wx      = (const float*)d_in[5];
  const float* Wh      = (const float*)d_in[6];
  const float* bvec    = (const float*)d_in[7];
  const float* Wxq     = (const float*)d_in[8];
  const float* Whq     = (const float*)d_in[9];
  const float* bq      = (const float*)d_in[10];
  const float* Wlf     = (const float*)d_in[11];
  const float* Ulf     = (const float*)d_in[12];
  const float* blf     = (const float*)d_in[13];
  const float* Wlb     = (const float*)d_in[14];
  const float* Ulb     = (const float*)d_in[15];
  const float* blb     = (const float*)d_in[16];
  const float* Wc      = (const float*)d_in[17];
  const float* bc      = (const float*)d_in[18];
  float* logits = (float*)d_out;

  // bump allocator over d_ws
  char* base = (char*)d_ws;
  size_t off = 0;
  auto alloc = [&](size_t bytes) -> void* {
    void* p = base + off;
    off = (off + bytes + 255) & ~(size_t)255;
    return p;
  };
  u16* Wxbf  = (u16*)alloc((size_t)kEp * kH * 2);     // row-major, K-padded
  u16* WhT   = (u16*)alloc((size_t)kH * kH * 2);      // transposed
  u16* WhqT  = (u16*)alloc((size_t)kH * kH * 2);      // transposed
  u16* Wxqbf = (u16*)alloc((size_t)kH * kH * 2);      // row-major
  u16* Wlfbf = (u16*)alloc((size_t)2 * kH * kG * 2);  // row-major
  u16* Wlbbf = (u16*)alloc((size_t)2 * kH * kG * 2);  // row-major
  u16* UlfT  = (u16*)alloc((size_t)kH * kG * 2);      // transposed (kG x kH)
  u16* UlbT  = (u16*)alloc((size_t)kH * kG * 2);      // transposed
  u16* x1bf  = (u16*)alloc((size_t)kBT * kEp * 2);
  u16* x2bf  = (u16*)alloc((size_t)kBT * kEp * 2);
  float* xw1 = (float*)alloc((size_t)kBT * kH * 4);
  float* xw2 = (float*)alloc((size_t)kBT * kH * 4);
  u16* e1bf  = (u16*)alloc((size_t)kBT * kH * 2);
  u16* e2bf  = (u16*)alloc((size_t)kBT * kH * 2);
  float* xwq = (float*)alloc((size_t)kBT * kH * 4);
  u16* e1qbf = (u16*)alloc((size_t)kBT * kH * 2);
  u16* encbf = (u16*)alloc((size_t)kBT * 512 * 2);
  float* xwf = (float*)alloc((size_t)kBT * kG * 4);
  float* xwb = (float*)alloc((size_t)kBT * kG * 4);
  float* hf  = (float*)alloc((size_t)kBT * kH * 4);
  float* hb  = (float*)alloc((size_t)kBT * kH * 4);

  auto cdiv = [](int a, int b) { return (a + b - 1) / b; };

  // 1) pack weights: GEMM weights row-major bf16, recurrence weights transposed
  pack_bf16<<<cdiv(kEp * kH, 256), 256, 0, stream>>>(Wx, Wxbf, kE, kH, kEp);
  pack_bf16<<<cdiv(kH * kH, 256), 256, 0, stream>>>(Wxq, Wxqbf, kH, kH, kH);
  pack_bf16<<<cdiv(2 * kH * kG, 256), 256, 0, stream>>>(Wlf, Wlfbf, 2 * kH, kG, 2 * kH);
  pack_bf16<<<cdiv(2 * kH * kG, 256), 256, 0, stream>>>(Wlb, Wlbbf, 2 * kH, kG, 2 * kH);
  pack_bf16_T<<<cdiv(kH * kH, 256), 256, 0, stream>>>(Wh, WhT, kH, kH);
  pack_bf16_T<<<cdiv(kH * kH, 256), 256, 0, stream>>>(Whq, WhqT, kH, kH);
  pack_bf16_T<<<cdiv(kH * kG, 256), 256, 0, stream>>>(Ulf, UlfT, kH, kG);
  pack_bf16_T<<<cdiv(kH * kG, 256), 256, 0, stream>>>(Ulb, UlbT, kH, kG);

  // 2) embedding gather -> bf16 (K padded to 320)
  gather_embed<<<cdiv(kBT * kEp, 256), 256, 0, stream>>>(tokens1, emb, x1bf);
  gather_embed<<<cdiv(kBT * kEp, 256), 256, 0, stream>>>(tokens2, emb, x2bf);

  // 3) text-encoder input projections: xw = x @ Wx + b
  {
    dim3 grid(kH / 16, kBT / 128);
    gemm_bf16<<<grid, 256, 0, stream>>>(x1bf, Wxbf, bvec, xw1, nullptr, kBT, kH, kEp,
                                        kEp, kH);
    gemm_bf16<<<grid, 256, 0, stream>>>(x2bf, Wxbf, bvec, xw2, nullptr, kBT, kH, kEp,
                                        kEp, kH);
  }

  // 4) shared tanh-RNN over both texts (grid.y selects sequence)
  {
    dim3 grid(kB / 16, 2);
    rnn_rec<<<grid, 256, 0, stream>>>(xw1, xw2, WhT, mask1, mask2, e1bf, e2bf);
  }

  // 5) question encoder: xwq = e1 @ Wxq + bq ; recurrence with WhqT
  {
    dim3 grid(kH / 16, kBT / 128);
    gemm_bf16<<<grid, 256, 0, stream>>>(e1bf, Wxqbf, bq, xwq, nullptr, kBT, kH, kH,
                                        kH, kH);
  }
  {
    dim3 grid(kB / 16, 1);
    rnn_rec<<<grid, 256, 0, stream>>>(xwq, xwq, WhqT, nullptr, nullptr, e1qbf,
                                      e1qbf);
  }

  // 6) fused coattention -> enc (bf16, B*T x 512)
  coattn<<<kB, 256, 0, stream>>>(e1qbf, e2bf, encbf);

  // 7) BiLSTM input projections: xw = enc @ W + b (big GEMMs, N=1024, K=512)
  {
    dim3 grid(kG / 16, kBT / 128);
    gemm_bf16<<<grid, 256, 0, stream>>>(encbf, Wlfbf, blf, xwf, nullptr, kBT, kG,
                                        2 * kH, 2 * kH, kG);
    gemm_bf16<<<grid, 256, 0, stream>>>(encbf, Wlbbf, blb, xwb, nullptr, kBT, kG,
                                        2 * kH, 2 * kH, kG);
  }

  // 8) LSTM recurrences, forward + backward in one grid
  {
    dim3 grid(kB / 16, 2);
    lstm_rec<<<grid, 256, 0, stream>>>(xwf, UlfT, hf, xwb, UlbT, hb);
  }

  // 9) classifier
  classify<<<kB, 256, 0, stream>>>(hf, hb, Wc, bc, logits);
}
